// THAN_ADJ_25958782337226
// MI455X (gfx1250) — matrix operand order repaired and checked by reading, not compiled, not run on hardware
//
#include <hip/hip_runtime.h>
#include <hip/hip_bf16.h>
#include <math.h>

#define B_     16
#define N_     1024
#define FIN_   80
#define KPAD_  96
#define D_     64
#define DEPTH_ 4
#define BN_EPS_ 1e-5f

typedef __attribute__((ext_vector_type(16))) _Float16 v16h;
typedef __attribute__((ext_vector_type(8)))  _Float16 v8h;
typedef __attribute__((ext_vector_type(8)))  float    v8f;

// Static device buffers (graph-replay safe; all fully rewritten each call).
__device__ float    g_h   [B_ * N_ * D_];      // node features f32
__device__ _Float16 g_h16 [B_ * N_ * D_];      // f16 copy (WMMA A operand source)
__device__ _Float16 g_hWT16[B_ * D_ * N_];     // relu(h@W) TRANSPOSED [b][d][j] (B operand source)
__device__ _Float16 g_X16 [B_ * N_ * KPAD_];   // X in f16, K zero-padded 80->96
__device__ _Float16 g_W0T [D_ * KPAD_];        // W0^T, K padded 80->96, [n][k]
__device__ _Float16 g_WlT [3 * D_ * D_];       // Wl^T per layer, [layer][n][k]
__device__ float    g_t1  [B_ * N_];
__device__ float    g_t2  [B_ * N_];
__device__ float    g_pool[B_ * D_];

static __device__ inline v8f zero8() {
  v8f z;
#pragma unroll
  for (int e = 0; e < 8; ++e) z[e] = 0.0f;
  return z;
}

// A operand (16x32 f16) from a row-major f16 row pointer already offset to
// [row][k0 + hi*8]: elements 0..7 = K k0+hi*8.., elements 8..15 = K k0+16+hi*8..
static __device__ inline v16h load_A_rows(const _Float16* __restrict__ p) {
  const v8h lo = *(const v8h*)p;
  const v8h hh = *(const v8h*)(p + 16);
  return __builtin_shufflevector(lo, hh, 0, 1, 2, 3, 4, 5, 6, 7,
                                 8, 9, 10, 11, 12, 13, 14, 15);
}

// B operand (32x16 f16): lane L -> col n=(L&15)+n0 ; elements 0..7 -> K=(L>>4)*8+e,
// elements 8..15 -> K=16+(L>>4)*8+(e-8). With a transposed ([n][k]) source: two 16B loads.
static __device__ inline v16h load_B_T(const _Float16* __restrict__ base) {
  const v8h lo = *(const v8h*)base;
  const v8h hi = *(const v8h*)(base + 16);
  return __builtin_shufflevector(lo, hi, 0, 1, 2, 3, 4, 5, 6, 7,
                                 8, 9, 10, 11, 12, 13, 14, 15);
}

static __device__ inline v8f wmma_f16(v16h a, v16h b, v8f c) {
  return __builtin_amdgcn_wmma_f32_16x16x32_f16(false, a, false, b, (short)0, c,
                                                false, false);
}

// ---- Prep A: f16-transposed weights ---------------------------------------
__global__ __launch_bounds__(256) void k_prep(const float* __restrict__ W0,
                                              const float* __restrict__ Wl) {
  const int i = blockIdx.x * 256 + threadIdx.x;
  if (i < D_ * KPAD_) {                 // W0^T with zero K-padding
    const int n = i / KPAD_, k = i % KPAD_;
    g_W0T[i] = (_Float16)((k < FIN_) ? W0[k * D_ + n] : 0.0f);
  }
  const int j = i - D_ * KPAD_;
  if (j >= 0 && j < 3 * D_ * D_) {      // Wl^T
    const int layer = j / (D_ * D_), rest = j % (D_ * D_);
    const int n = rest / D_, k = rest % D_;
    g_WlT[j] = (_Float16)Wl[(layer * D_ + k) * D_ + n];
  }
}

// ---- Prep B: zero-padded f16 copy of X (streaming, coalesced) -------------
__global__ __launch_bounds__(256) void k_prepx(const float* __restrict__ X) {
  const int i = blockIdx.x * 256 + threadIdx.x;    // 0 .. B*N*KPAD-1
  const int r = i / KPAD_, k = i - r * KPAD_;
  g_X16[i] = (_Float16)((k < FIN_) ? X[(size_t)r * FIN_ + k] : 0.0f);
}

// ---- Kernel 1: h = relu(X @ W0). Same shape as layer GEMM, no guards ------
__global__ __launch_bounds__(32) void k_in_gemm() {
  const int lane = threadIdx.x;
  const int row0 = blockIdx.x * 16;
  const int m = lane & 15, hi = lane >> 4;
  v8f c[4];
#pragma unroll
  for (int t = 0; t < 4; ++t) c[t] = zero8();

#pragma unroll
  for (int kt = 0; kt < 3; ++kt) {            // K = 80 padded to 96
    const int k0 = kt * 32;
    const v16h a = load_A_rows(g_X16 + (size_t)(row0 + m) * KPAD_ + k0 + (hi << 3));
#pragma unroll
    for (int t = 0; t < 4; ++t) {
      const int n = t * 16 + m;
      const v16h b = load_B_T(g_W0T + n * KPAD_ + k0 + (hi << 3));
      c[t] = wmma_f16(a, b, c[t]);
    }
  }
#pragma unroll
  for (int r = 0; r < 8; ++r)
#pragma unroll
    for (int t = 0; t < 4; ++t) {
      const int rowg = row0 + r + 8 * hi, col = t * 16 + m;
      float v = c[t][r]; v = v > 0.0f ? v : 0.0f;
      const size_t idx = (size_t)rowg * D_ + col;
      g_h[idx] = v; g_h16[idx] = (_Float16)v;
    }
}

// ---- Kernel 2: hW^T = relu(h @ Wl[i])^T (stored [b][d][j]) ----------------
__global__ __launch_bounds__(32) void k_layer_gemm(int layer) {
  const int lane = threadIdx.x;
  const int row0 = blockIdx.x * 16;
  const int m = lane & 15, hi = lane >> 4;
  const _Float16* WT = g_WlT + layer * D_ * D_;
  v8f c[4];
#pragma unroll
  for (int t = 0; t < 4; ++t) c[t] = zero8();

#pragma unroll
  for (int kt = 0; kt < 2; ++kt) {            // K = 64
    const int k0 = kt * 32;
    const v16h a = load_A_rows(g_h16 + (size_t)(row0 + m) * D_ + k0 + (hi << 3));
#pragma unroll
    for (int t = 0; t < 4; ++t) {
      const int n = t * 16 + m;
      const v16h b = load_B_T(WT + n * D_ + k0 + (hi << 3));
      c[t] = wmma_f16(a, b, c[t]);
    }
  }
#pragma unroll
  for (int r = 0; r < 8; ++r)
#pragma unroll
    for (int t = 0; t < 4; ++t) {
      const int rowg = row0 + r + 8 * hi, col = t * 16 + m;
      float v = c[t][r]; v = v > 0.0f ? v : 0.0f;
      // transposed store: [b][d=col][j = rowg%N]
      g_hWT16[(((size_t)(rowg >> 10) * D_ + col) << 10) + (rowg & 1023)] = (_Float16)v;
    }
}

// ---- Kernel 3: t1/t2 per node (coalesced reads of hW^T) -------------------
__global__ __launch_bounds__(256) void k_t12(const float* __restrict__ a1,
                                             const float* __restrict__ a2) {
  const int idx = blockIdx.x * 256 + threadIdx.x;   // 0 .. B*N-1
  const int b = idx >> 10, n = idx & 1023;
  float s1 = 0.0f, s2 = 0.0f;
#pragma unroll 8
  for (int d = 0; d < D_; ++d) {
    const float v = (float)g_hWT16[(((size_t)(b * D_ + d)) << 10) + n];
    s1 += v * a1[d]; s2 += v * a2[d];
  }
  g_t1[idx] = s1; g_t2[idx] = s2;
}

// ---- Kernel 4: fused masked-attention (flash style) + root gate -----------
// A (prob tile) built fully in registers; B (hW^T tile) two 16B global loads.
__global__ __launch_bounds__(32) void k_attn(const int* __restrict__ adj,
                                             const float* __restrict__ root_list,
                                             int root_row) {
  const int lane = threadIdx.x;
  const int b    = blockIdx.x >> 6;
  const int i0   = (blockIdx.x & 63) * 16;
  const int m    = lane & 15;      // score row (and output col index)
  const int hi   = lane >> 4;

  const float t1v = g_t1[b * N_ + i0 + m];
  const int*   arow = adj  + ((size_t)b * N_ + i0 + m) * N_;
  const float* t2p  = g_t2 + b * N_;
  float mrun = 0.0f, s = 0.0f;     // e >= 0 always, so m=0 start is exact
  v8f acc[4];
#pragma unroll
  for (int t = 0; t < 4; ++t) acc[t] = zero8();

  for (int jt = 0; jt < N_ / 32; ++jt) {
    const int jb = jt * 32;
    if (jt + 2 < N_ / 32)                       // gfx1250 global_prefetch_b8
      __builtin_prefetch(arow + jb + 64, 0, 1);

    float ev[16];
    float tmax = 0.0f;
#pragma unroll
    for (int e = 0; e < 16; ++e) {              // K in A-operand order
      const int k = jb + ((e >> 3) << 4) + (hi << 3) + (e & 7);
      float x = t1v + t2p[k];
      x = x > 0.0f ? x : 0.0f;
      x = (arow[k] > 0) ? x : 0.0f;
      ev[e] = x;
      tmax = fmaxf(tmax, x);
    }
    tmax = fmaxf(tmax, __shfl_xor(tmax, 16));   // lanes L, L^16 cover all 32 j
    const float m_new = fmaxf(mrun, tmax);
    const float scale = __expf(mrun - m_new);
    float psum = 0.0f;
    v16h a;
#pragma unroll
    for (int e = 0; e < 16; ++e) {
      const float p = __expf(ev[e] - m_new);
      psum += p;
      a[e] = (_Float16)p;                       // A operand built in registers
    }
    psum += __shfl_xor(psum, 16);
    s = s * scale + psum;
    mrun = m_new;

#pragma unroll
    for (int r = 0; r < 8; ++r) {               // rescale acc rows r+8*hi
      const float f = __shfl(scale, r + 8 * hi);
#pragma unroll
      for (int t = 0; t < 4; ++t) acc[t][r] = acc[t][r] * f;
    }
#pragma unroll
    for (int t = 0; t < 4; ++t) {
      const _Float16* vp = g_hWT16 +
          (((size_t)(b * D_ + t * 16 + m)) << 10) + jb + (hi << 3);
      acc[t] = wmma_f16(a, load_B_T(vp), acc[t]);
    }
  }

  const float sinv = 1.0f / s;
#pragma unroll
  for (int r = 0; r < 8; ++r) {
    const int rowg = i0 + r + 8 * hi;
    const float inv = __shfl(sinv, r + 8 * hi);
    const float rv = root_list[((size_t)b * DEPTH_ + root_row) * N_ + rowg];
#pragma unroll
    for (int t = 0; t < 4; ++t) {
      const int col = t * 16 + m;
      const size_t idx = ((size_t)b * N_ + rowg) * D_ + col;
      const float emb = acc[t][r] * inv;
      const float hv = g_h[idx];                // unconditional load -> cndmask
      const float v = (rv > 0.0f) ? emb : hv;
      g_h[idx] = v; g_h16[idx] = (_Float16)v;
    }
  }
}

// ---- Kernel 5: masked-softmax readout -> g_pool (16,64) -------------------
__global__ __launch_bounds__(256) void k_readout(const float* __restrict__ P,
                                                 const float* __restrict__ root_list) {
  __shared__ float se[N_];
  __shared__ float sred[256];
  const int b = blockIdx.x, tid = threadIdx.x;

  float lmax = -3.4e38f;
  for (int q = 0; q < 4; ++q) {
    const int n = tid + q * 256;
    const float* hp = g_h + ((size_t)b * N_ + n) * D_;
    float dot = 0.0f;
#pragma unroll 8
    for (int d = 0; d < D_; ++d) dot += hp[d] * P[d];
    dot = dot > 0.0f ? dot : 0.0f;
    const float rv = root_list[((size_t)b * DEPTH_ + 1) * N_ + n];
    const float e = (rv > 0.0f) ? dot : -9.0e15f;
    se[n] = e; lmax = fmaxf(lmax, e);
  }
  sred[tid] = lmax; __syncthreads();
  for (int off = 128; off; off >>= 1) {
    if (tid < off) sred[tid] = fmaxf(sred[tid], sred[tid + off]);
    __syncthreads();
  }
  const float mx = sred[0]; __syncthreads();

  float lsum = 0.0f;
  for (int q = 0; q < 4; ++q) {
    const int n = tid + q * 256;
    const float p = __expf(se[n] - mx);
    se[n] = p; lsum += p;
  }
  sred[tid] = lsum; __syncthreads();
  for (int off = 128; off; off >>= 1) {
    if (tid < off) sred[tid] += sred[tid + off];
    __syncthreads();
  }
  const float inv = 1.0f / sred[0]; __syncthreads();

  const int d = tid & 63, part = tid >> 6;
  float acc = 0.0f;
  for (int q = 0; q < 256; ++q) {
    const int n = part * 256 + q;
    acc += se[n] * g_h[((size_t)b * N_ + n) * D_ + d];
  }
  sred[tid] = acc * inv; __syncthreads();
  if (tid < 64)
    g_pool[b * D_ + tid] = sred[tid] + sred[tid + 64] + sred[tid + 128] + sred[tid + 192];
}

// ---- Kernel 6: MLP head with frozen BN + softmax --------------------------
__global__ __launch_bounds__(256) void k_mlp(
    const float* __restrict__ pW1, const float* __restrict__ pb1,
    const float* __restrict__ g1,  const float* __restrict__ be1,
    const float* __restrict__ m1,  const float* __restrict__ v1,
    const float* __restrict__ pW2, const float* __restrict__ pb2,
    const float* __restrict__ g2,  const float* __restrict__ be2,
    const float* __restrict__ m2,  const float* __restrict__ v2,
    const float* __restrict__ pW3, const float* __restrict__ pb3,
    float* __restrict__ out) {
  __shared__ float x0[B_ * 64];
  __shared__ float x1[B_ * 128];
  __shared__ float x2[B_ * 64];
  __shared__ float lg[B_ * 2];
  const int tid = threadIdx.x;

  for (int i = tid; i < B_ * 64; i += 256) x0[i] = g_pool[i];
  __syncthreads();
  for (int o = tid; o < B_ * 128; o += 256) {
    const int r = o >> 7, c = o & 127;
    float acc = pb1[c];
#pragma unroll 8
    for (int d = 0; d < 64; ++d) acc += x0[r * 64 + d] * pW1[d * 128 + c];
    acc = (acc - m1[c]) * rsqrtf(v1[c] + BN_EPS_) * g1[c] + be1[c];
    x1[o] = acc > 0.0f ? acc : 0.0f;
  }
  __syncthreads();
  for (int o = tid; o < B_ * 64; o += 256) {
    const int r = o >> 6, c = o & 63;
    float acc = pb2[c];
#pragma unroll 8
    for (int d = 0; d < 128; ++d) acc += x1[r * 128 + d] * pW2[d * 64 + c];
    acc = (acc - m2[c]) * rsqrtf(v2[c] + BN_EPS_) * g2[c] + be2[c];
    x2[o] = acc > 0.0f ? acc : 0.0f;
  }
  __syncthreads();
  if (tid < B_ * 2) {
    const int r = tid >> 1, c = tid & 1;
    float acc = pb3[c];
#pragma unroll 8
    for (int d = 0; d < 64; ++d) acc += x2[r * 64 + d] * pW3[d * 2 + c];
    lg[tid] = acc > 0.0f ? acc : 0.0f;
  }
  __syncthreads();
  if (tid < B_) {
    const float l0 = lg[tid * 2], l1 = lg[tid * 2 + 1];
    const float mx = fmaxf(l0, l1);
    const float e0 = __expf(l0 - mx), e1 = __expf(l1 - mx);
    const float z = e0 + e1;
    out[tid * 2 + 0] = e0 / z;
    out[tid * 2 + 1] = e1 / z;
  }
}

extern "C" void kernel_launch(void* const* d_in, const int* in_sizes, int n_in,
                              void* d_out, int out_size, void* d_ws, size_t ws_size,
                              hipStream_t stream) {
  (void)in_sizes; (void)n_in; (void)out_size; (void)d_ws; (void)ws_size;
  const int*   adj  = (const int*)  d_in[0];
  const float* X    = (const float*)d_in[1];
  const float* root = (const float*)d_in[2];
  const float* W0   = (const float*)d_in[3];
  const float* Wl   = (const float*)d_in[4];
  const float* a1l  = (const float*)d_in[5];
  const float* a2l  = (const float*)d_in[6];
  const float* P    = (const float*)d_in[7];
  const float* pW1  = (const float*)d_in[8];
  const float* pb1  = (const float*)d_in[9];
  const float* g1   = (const float*)d_in[10];
  const float* be1  = (const float*)d_in[11];
  const float* m1   = (const float*)d_in[12];
  const float* v1   = (const float*)d_in[13];
  const float* pW2  = (const float*)d_in[14];
  const float* pb2  = (const float*)d_in[15];
  const float* g2   = (const float*)d_in[16];
  const float* be2  = (const float*)d_in[17];
  const float* m2   = (const float*)d_in[18];
  const float* v2   = (const float*)d_in[19];
  const float* pW3  = (const float*)d_in[20];
  const float* pb3  = (const float*)d_in[21];

  const int prep_elems = D_ * KPAD_ + 3 * D_ * D_;
  k_prep<<<(prep_elems + 255) / 256, 256, 0, stream>>>(W0, Wl);
  k_prepx<<<(B_ * N_ * KPAD_) / 256, 256, 0, stream>>>(X);
  k_in_gemm<<<(B_ * N_) / 16, 32, 0, stream>>>();
  for (int layer = 0; layer < DEPTH_ - 1; ++layer) {
    k_layer_gemm<<<(B_ * N_) / 16, 32, 0, stream>>>(layer);
    k_t12<<<(B_ * N_) / 256, 256, 0, stream>>>(a1l + layer * D_, a2l + layer * D_);
    k_attn<<<B_ * (N_ / 16), 32, 0, stream>>>(adj, root, DEPTH_ - 1 - layer);
  }
  k_readout<<<B_, 256, 0, stream>>>(P, root);
  k_mlp<<<1, 256, 0, stream>>>(pW1, pb1, g1, be1, m1, v1,
                               pW2, pb2, g2, be2, m2, v2,
                               pW3, pb3, (float*)d_out);
}
